// SelfAttention_50216757624841
// MI455X (gfx1250) — matrix operand order repaired and checked by reading, not compiled, not run on hardware
//
#include <hip/hip_runtime.h>
#include <hip/hip_bf16.h>

#define D_MODEL 1024
#define N_HEADS 16
#define HEAD_DIM 64
#define SEQ 2048
#define BATCH 4
#define M_TOTAL (BATCH * SEQ) // 8192
#define K_CHUNK 64

typedef unsigned int uint;
typedef unsigned short ushort;
typedef __attribute__((ext_vector_type(4)))  uint   u32x4;
typedef __attribute__((ext_vector_type(4)))  float  f32x4;
typedef __attribute__((ext_vector_type(16))) __bf16 v16bf;
typedef __attribute__((ext_vector_type(8)))  float  v8f;

union Frag {
  uint  u[8];
  u32x4 q[2];
  v16bf bf;
};

__device__ __forceinline__ ushort f2bf(float f) {
  uint u = __float_as_uint(f);
  u += 0x7FFFu + ((u >> 16) & 1u);   // round-to-nearest-even
  return (ushort)(u >> 16);
}
__device__ __forceinline__ uint pk2(float lo, float hi) {
  return (uint)f2bf(lo) | ((uint)f2bf(hi) << 16);
}
// wave-relative LDS byte offset for async-to-LDS VDST operand
__device__ __forceinline__ uint ldsOff(const void* p) {
  return (uint)(unsigned long long)(__attribute__((address_space(3))) const void*)p;
}
__device__ __forceinline__ void asyncWait0() {
  asm volatile("s_wait_asynccnt 0x0" ::: "memory");
}
// DPP16 ROW_XMASK cross-lane xor within a 16-lane row (wave32 half)
template <int MASK>
__device__ __forceinline__ float dpp_xor(float v) {
  return __uint_as_float((uint)__builtin_amdgcn_mov_dpp(
      (int)__float_as_uint(v), 0x160 | MASK, 0xf, 0xf, true));
}
__device__ __forceinline__ float rowMax16(float v) {
  v = fmaxf(v, dpp_xor<1>(v));
  v = fmaxf(v, dpp_xor<2>(v));
  v = fmaxf(v, dpp_xor<4>(v));
  v = fmaxf(v, dpp_xor<8>(v));
  return v;
}
__device__ __forceinline__ float rowSum16(float v) {
  v += dpp_xor<1>(v);
  v += dpp_xor<2>(v);
  v += dpp_xor<4>(v);
  v += dpp_xor<8>(v);
  return v;
}

// 16-bit A-matrix fragment K pattern (CDNA5 ISA 7.12.2): VGPR g holds K pair
// at 2g (g<4) or 16+2(g-4) (g>=4); upper half-wave adds +8.
__device__ __forceinline__ int kpatA(int g) { return (g < 4) ? 2 * g : 8 + 2 * g; }

// ---------------------------------------------------------------------------
// GEMM: C[8192,1024] = A @ W + bias, times scale. Block 256 thr = 8 waves;
// tile 128(M) x 64(N); wave tile 32x32; K-chunk 64; double-buffered LDS so
// chunk t+1 global traffic overlaps chunk t WMMAs. bf16-A path streams the
// A tile with global_load_async_to_lds_b128 (ASYNCcnt DMA, no VGPR bounce).
// ---------------------------------------------------------------------------
template <typename AT, bool BF16_OUT, bool HEAD_STORE>
__global__ __launch_bounds__(256) void gemm_bias_kernel(
    const AT* __restrict__ A, const float* __restrict__ W,
    const float* __restrict__ bias, void* __restrict__ Cout, float scale)
{
  constexpr bool ISBF = (sizeof(AT) == 2);
  __shared__ ushort As[2][128 * 64];  // [m][k] bf16  (2 x 16 KB)
  __shared__ ushort Bs[2][64 * 64];   // [n][k] bf16  (2 x 8 KB, W transposed)

  const int tid  = threadIdx.x;
  const int wave = tid >> 5;
  const int lane = tid & 31;
  const int wm = wave & 3;
  const int wn = wave >> 2;
  const int lrow   = lane & 15;
  const int koff8  = (lane >= 16) ? 8 : 0;
  const int kb16   = (lane >= 16) ? 16 : 0;
  const int rhalf8 = (lane >= 16) ? 8 : 0;

  const int blockN = blockIdx.x * 64;
  const int blockM = blockIdx.y * 128;

  const int arow = tid >> 1, ahalf = tid & 1;      // A: 128 rows x 2 halves
  const int bn = tid & 63, bkp = (tid >> 6) * 16;  // W: 64 cols x 4 k-groups

  struct WRegs { float w[16]; };
  struct ARegs { f32x4 a[8]; };

  auto loadW = [&](int k0) {
    WRegs r;
    const float* wp = W + (size_t)(k0 + bkp) * D_MODEL + blockN + bn;
#pragma unroll
    for (int i = 0; i < 16; ++i) r.w[i] = wp[(size_t)i * D_MODEL];
    return r;
  };
  auto storeW = [&](const WRegs& r, int buf) {
    u32x4 p0 = { pk2(r.w[0], r.w[1]),   pk2(r.w[2], r.w[3]),
                 pk2(r.w[4], r.w[5]),   pk2(r.w[6], r.w[7]) };
    u32x4 p1 = { pk2(r.w[8], r.w[9]),   pk2(r.w[10], r.w[11]),
                 pk2(r.w[12], r.w[13]), pk2(r.w[14], r.w[15]) };
    *(u32x4*)&Bs[buf][bn * 64 + bkp]     = p0;
    *(u32x4*)&Bs[buf][bn * 64 + bkp + 8] = p1;
  };
  auto loadAf = [&](int k0) {  // fp32 A -> registers
    ARegs r;
    const f32x4* ap4 =
        (const f32x4*)(A + (size_t)(blockM + arow) * D_MODEL + k0 + ahalf * 32);
#pragma unroll
    for (int i = 0; i < 8; ++i) r.a[i] = ap4[i];
    return r;
  };
  auto storeAf = [&](const ARegs& r, int buf) {  // convert + pack to LDS
    ushort* ad = &As[buf][arow * 64 + ahalf * 32];
#pragma unroll
    for (int v = 0; v < 4; ++v) {
      const f32x4 a0 = r.a[2 * v], a1 = r.a[2 * v + 1];
      u32x4 pk = { pk2(a0.x, a0.y), pk2(a0.z, a0.w), pk2(a1.x, a1.y), pk2(a1.z, a1.w) };
      *(u32x4*)(ad + v * 8) = pk;
    }
  };
  auto stageAasync = [&](int k0, int buf) {  // bf16 A -> LDS via async DMA
    const unsigned long long g =
        (unsigned long long)(A + (size_t)(blockM + arow) * D_MODEL + k0 + ahalf * 32);
    const uint l = ldsOff(&As[buf][arow * 64 + ahalf * 32]);
    asm volatile("global_load_async_to_lds_b128 %0, %1, off"           :: "v"(l), "v"(g) : "memory");
    asm volatile("global_load_async_to_lds_b128 %0, %1, off offset:16" :: "v"(l), "v"(g) : "memory");
    asm volatile("global_load_async_to_lds_b128 %0, %1, off offset:32" :: "v"(l), "v"(g) : "memory");
    asm volatile("global_load_async_to_lds_b128 %0, %1, off offset:48" :: "v"(l), "v"(g) : "memory");
  };

  v8f zero8;
#pragma unroll
  for (int i = 0; i < 8; ++i) zero8[i] = 0.0f;
  v8f acc[2][2];
#pragma unroll
  for (int mt = 0; mt < 2; ++mt)
#pragma unroll
    for (int nt = 0; nt < 2; ++nt) acc[mt][nt] = zero8;

  // prologue: stage chunk 0 into buffer 0
  ARegs ar;
  WRegs wr = loadW(0);
  if constexpr (ISBF) stageAasync(0, 0);
  else { ar = loadAf(0); storeAf(ar, 0); }
  storeW(wr, 0);
  if constexpr (ISBF) asyncWait0();
  __syncthreads();

  int cur = 0;
  for (int k0 = 0; k0 < D_MODEL; k0 += K_CHUNK) {
    const bool more = (k0 + K_CHUNK) < D_MODEL;  // scalar-uniform
    const int nxt = cur ^ 1;
    if (more) {  // issue next chunk's global traffic before computing
      if constexpr (ISBF) stageAasync(k0 + K_CHUNK, nxt);
      else ar = loadAf(k0 + K_CHUNK);
      wr = loadW(k0 + K_CHUNK);
    }

#pragma unroll
    for (int ks = 0; ks < 2; ++ks) {
      Frag af[2], bfg[2];
#pragma unroll
      for (int mt = 0; mt < 2; ++mt)
#pragma unroll
        for (int g = 0; g < 8; ++g)
          af[mt].u[g] = *(const uint*)&As[cur]
              [(wm * 32 + mt * 16 + lrow) * 64 + ks * 32 + kpatA(g) + koff8];
#pragma unroll
      for (int nt = 0; nt < 2; ++nt) {
        const u32x4* bp =
            (const u32x4*)&Bs[cur][(wn * 32 + nt * 16 + lrow) * 64 + ks * 32 + koff8];
        bfg[nt].q[0] = bp[0];
        bfg[nt].q[1] = bp[2];
      }
#pragma unroll
      for (int mt = 0; mt < 2; ++mt)
#pragma unroll
        for (int nt = 0; nt < 2; ++nt)
          acc[mt][nt] = __builtin_amdgcn_wmma_f32_16x16x32_bf16(
              false, af[mt].bf, false, bfg[nt].bf, (short)0, acc[mt][nt], false, false);
    }

    if (more) {
      if constexpr (!ISBF) storeAf(ar, nxt);
      storeW(wr, nxt);
    }
    if constexpr (ISBF) asyncWait0();
    __syncthreads();
    cur = nxt;
  }

#pragma unroll
  for (int mt = 0; mt < 2; ++mt)
#pragma unroll
    for (int nt = 0; nt < 2; ++nt) {
      const int n = blockN + wn * 32 + nt * 16 + lrow;
      const float bv = bias[n];
#pragma unroll
      for (int g = 0; g < 8; ++g) {
        const int m = blockM + wm * 32 + mt * 16 + g + rhalf8;
        const float val = (acc[mt][nt][g] + bv) * scale;
        size_t idx;
        if constexpr (HEAD_STORE) {
          const int b = m >> 11, s = m & (SEQ - 1);
          const int h = n >> 6, d = n & 63;
          idx = ((size_t)(b * N_HEADS + h) * SEQ + s) * HEAD_DIM + d;
        } else {
          idx = (size_t)m * D_MODEL + n;
        }
        if constexpr (BF16_OUT) ((ushort*)Cout)[idx] = f2bf(val);
        else                    ((float*)Cout)[idx]  = val;
      }
    }
}

// ---------------------------------------------------------------------------
// Flash attention, bf16 Q,K,V in [B,H,S,hd]; Q pre-scaled by log2(e)/sqrt(hd)
// so softmax runs on v_exp_f32 (2^x) directly. Double-buffered K/V tiles;
// K staged with async-to-LDS DMA; softmax reductions use DPP16 ROW_XMASK.
// ---------------------------------------------------------------------------
__global__ __launch_bounds__(256) void attn_kernel(
    const ushort* __restrict__ Qb, const ushort* __restrict__ Kb,
    const ushort* __restrict__ Vb, ushort* __restrict__ Ob)
{
  __shared__ ushort Qs[128 * 64];       // [qrow][dim]          16 KB
  __shared__ ushort Ks[2][64 * 64];     // [key][dim] x2        16 KB
  __shared__ ushort Vs[2][64 * 64];     // [dim][key] x2        16 KB
  __shared__ ushort Ps[8 * 16 * 64];    // per-wave P scratch   16 KB

  const int tid  = threadIdx.x;
  const int wave = tid >> 5;
  const int lane = tid & 31;
  const int lrow   = lane & 15;
  const int koff8  = (lane >= 16) ? 8 : 0;
  const int kb16   = (lane >= 16) ? 16 : 0;
  const int rhalf8 = (lane >= 16) ? 8 : 0;

  const int bh   = blockIdx.x;   // b*16 + h
  const int qblk = blockIdx.y;
  const size_t headBase = (size_t)bh * SEQ * HEAD_DIM;

  const int row = tid >> 2, part = tid & 3;  // K/V staging: 64 rows x 4 parts

  struct VRegs { u32x4 q0, q1; };
  auto stageK = [&](int kt, int buf) {
    const unsigned long long g = (unsigned long long)(
        Kb + headBase + (size_t)(kt * 64 + row) * 64 + part * 16);
    const uint l = ldsOff(&Ks[buf][row * 64 + part * 16]);
    asm volatile("global_load_async_to_lds_b128 %0, %1, off"           :: "v"(l), "v"(g) : "memory");
    asm volatile("global_load_async_to_lds_b128 %0, %1, off offset:16" :: "v"(l), "v"(g) : "memory");
  };
  auto loadV = [&](int kt) {
    const u32x4* vs4 = (const u32x4*)(
        Vb + headBase + (size_t)(kt * 64 + row) * 64 + part * 16);
    VRegs r;
    r.q0 = vs4[0];
    r.q1 = vs4[1];
    return r;
  };
  auto storeV = [&](VRegs r, int buf) {  // transpose into [dim][key]
    union { u32x4 q[2]; ushort s[16]; } vt;
    vt.q[0] = r.q0;
    vt.q[1] = r.q1;
#pragma unroll
    for (int i = 0; i < 16; ++i) Vs[buf][(part * 16 + i) * 64 + row] = vt.s[i];
  };

  {  // stage Q tile (once) + tile 0 of K/V, all behind one barrier
    const int qrow = tid >> 1, qpart = tid & 1;
    const u32x4* src =
        (const u32x4*)(Qb + headBase + (size_t)(qblk * 128 + qrow) * 64 + qpart * 32);
    u32x4* dst = (u32x4*)&Qs[qrow * 64 + qpart * 32];
#pragma unroll
    for (int i = 0; i < 4; ++i) dst[i] = src[i];
    stageK(0, 0);
    storeV(loadV(0), 0);
    asyncWait0();
  }
  __syncthreads();

  Frag qa[2];
#pragma unroll
  for (int ks = 0; ks < 2; ++ks)
#pragma unroll
    for (int g = 0; g < 8; ++g)
      qa[ks].u[g] = *(const uint*)&Qs[(wave * 16 + lrow) * 64 + ks * 32 + kpatA(g) + koff8];

  v8f zero8;
#pragma unroll
  for (int i = 0; i < 8; ++i) zero8[i] = 0.0f;
  v8f acc[4];
#pragma unroll
  for (int dt = 0; dt < 4; ++dt) acc[dt] = zero8;
  float mrun[8], lrun[8];
#pragma unroll
  for (int g = 0; g < 8; ++g) { mrun[g] = -1e30f; lrun[g] = 0.0f; }

  int cur = 0;
  for (int kt = 0; kt < SEQ / 64; ++kt) {
    const int nxt = cur ^ 1;
    const bool more = (kt + 1) < (SEQ / 64);  // scalar-uniform
    VRegs vr;
    if (more) {         // prefetch next tile while computing on current
      stageK(kt + 1, nxt);
      vr = loadV(kt + 1);
    }

    // scores: S = Q(16x64) @ K^T
    v8f sc[4];
#pragma unroll
    for (int ns = 0; ns < 4; ++ns) {
      sc[ns] = zero8;
#pragma unroll
      for (int ks = 0; ks < 2; ++ks) {
        Frag kf;
        const u32x4* kp =
            (const u32x4*)&Ks[cur][(ns * 16 + lrow) * 64 + ks * 32 + koff8];
        kf.q[0] = kp[0];
        kf.q[1] = kp[2];
        sc[ns] = __builtin_amdgcn_wmma_f32_16x16x32_bf16(
            false, qa[ks].bf, false, kf.bf, (short)0, sc[ns], false, false);
      }
    }

    // online softmax (base-2); DPP ROW_XMASK reductions stay in each 16-lane half
#pragma unroll
    for (int g = 0; g < 8; ++g) {
      const float s0 = sc[0][g], s1 = sc[1][g], s2 = sc[2][g], s3 = sc[3][g];
      const float vmax = rowMax16(fmaxf(fmaxf(s0, s1), fmaxf(s2, s3)));
      const float mnew = fmaxf(mrun[g], vmax);
      const float corr = __builtin_amdgcn_exp2f(mrun[g] - mnew);
      const float p0 = __builtin_amdgcn_exp2f(s0 - mnew);
      const float p1 = __builtin_amdgcn_exp2f(s1 - mnew);
      const float p2 = __builtin_amdgcn_exp2f(s2 - mnew);
      const float p3 = __builtin_amdgcn_exp2f(s3 - mnew);
      const float rs = rowSum16((p0 + p1) + (p2 + p3));
      lrun[g] = lrun[g] * corr + rs;
      mrun[g] = mnew;
      acc[0][g] *= corr; acc[1][g] *= corr; acc[2][g] *= corr; acc[3][g] *= corr;
      ushort* pw = &Ps[wave * 1024 + (g + rhalf8) * 64 + lrow];
      pw[0] = f2bf(p0); pw[16] = f2bf(p1); pw[32] = f2bf(p2); pw[48] = f2bf(p3);
    }

    // O += P(16x64) @ V
    Frag pa[2];
#pragma unroll
    for (int ks = 0; ks < 2; ++ks)
#pragma unroll
      for (int g = 0; g < 8; ++g)
        pa[ks].u[g] =
            *(const uint*)&Ps[wave * 1024 + lrow * 64 + ks * 32 + kpatA(g) + koff8];

#pragma unroll
    for (int dt = 0; dt < 4; ++dt)
#pragma unroll
      for (int ks = 0; ks < 2; ++ks) {
        Frag vf;
        const u32x4* vp =
            (const u32x4*)&Vs[cur][(dt * 16 + lrow) * 64 + ks * 32 + koff8];
        vf.q[0] = vp[0];
        vf.q[1] = vp[2];
        acc[dt] = __builtin_amdgcn_wmma_f32_16x16x32_bf16(
            false, pa[ks].bf, false, vf.bf, (short)0, acc[dt], false, false);
      }

    if (more) storeV(vr, nxt);
    asyncWait0();
    __syncthreads();
    cur = nxt;
  }

  // normalize and store to [B,S,D] bf16
  const int b = bh >> 4, h = bh & 15;
#pragma unroll
  for (int g = 0; g < 8; ++g) {
    const float inv = __builtin_amdgcn_rcpf(lrun[g]);
    const int s = qblk * 128 + wave * 16 + g + rhalf8;
    ushort* orow = Ob + ((size_t)(b * SEQ + s)) * D_MODEL + h * HEAD_DIM + lrow;
#pragma unroll
    for (int dt = 0; dt < 4; ++dt) orow[dt * 16] = f2bf(acc[dt][g] * inv);
  }
}

// ---------------------------------------------------------------------------
extern "C" void kernel_launch(void* const* d_in, const int* in_sizes, int n_in,
                              void* d_out, int out_size, void* d_ws, size_t ws_size,
                              hipStream_t stream)
{
  const float* x  = (const float*)d_in[0];
  const float* Wq = (const float*)d_in[1];
  const float* bq = (const float*)d_in[2];
  const float* Wk = (const float*)d_in[3];
  const float* bk = (const float*)d_in[4];
  const float* Wv = (const float*)d_in[5];
  const float* bv = (const float*)d_in[6];
  const float* Wo = (const float*)d_in[7];
  const float* bo = (const float*)d_in[8];

  const size_t HE = (size_t)BATCH * N_HEADS * SEQ * HEAD_DIM;  // 8388608 elems
  ushort* Qb = (ushort*)d_ws;  // bf16 buffers: 4 x 16 MB = 64 MB total
  ushort* Kb = Qb + HE;
  ushort* Vb = Kb + HE;
  ushort* Ob = Vb + HE;

  dim3 tb(256);
  dim3 gg(D_MODEL / 64, M_TOTAL / 128);  // (16, 64)
  // 1/sqrt(hd) * log2(e): scores land pre-scaled for base-2 softmax
  const float qscale = 0.125f * 1.4426950408889634f;

  gemm_bias_kernel<float,  true,  true ><<<gg, tb, 0, stream>>>(x, Wq, bq, (void*)Qb, qscale);
  gemm_bias_kernel<float,  true,  true ><<<gg, tb, 0, stream>>>(x, Wk, bk, (void*)Kb, 1.0f);
  gemm_bias_kernel<float,  true,  true ><<<gg, tb, 0, stream>>>(x, Wv, bv, (void*)Vb, 1.0f);

  attn_kernel<<<dim3(BATCH * N_HEADS, SEQ / 128), tb, 0, stream>>>(Qb, Kb, Vb, Ob);

  gemm_bias_kernel<ushort, false, false><<<gg, tb, 0, stream>>>(Ob, Wo, bo, d_out, 1.0f);
}
